// Attention_53463752901338
// MI455X (gfx1250) — compile-verified, matrix operand order checked by reading, not run on hardware
//
#include <hip/hip_runtime.h>
#include <hip/hip_bf16.h>

typedef __attribute__((ext_vector_type(16))) __bf16 v16bf;
typedef __attribute__((ext_vector_type(8)))  float  v8f;

#define BATCH 16
#define SEQ   1024
#define DIM   2048
#define MTOT  (BATCH*SEQ)      // 16384
#define BM    128
#define BN    256
#define KC    64               // k-chunk staged in LDS per iteration
#define NSTG  (DIM/KC)         // 32
#define LDA   80               // bf16 elems per LDS row (64 + 16 pad -> 160B, 32B aligned)

__device__ __forceinline__ unsigned bf16bits(float f) {
    union { float f; unsigned u; } v; v.f = f;
    unsigned r = v.u + 0x7FFFu + ((v.u >> 16) & 1u);   // round-to-nearest-even
    return r >> 16;
}
__device__ __forceinline__ unsigned pack2(float a, float b) {
    return bf16bits(a) | (bf16bits(b) << 16);
}

// Load one 16x32 bf16 fragment's per-lane data (16 bf16 = two 16B chunks):
// lanes 0-15 hold K={0..7,16..23}, lanes 16-31 hold K={8..15,24..31}.
__device__ __forceinline__ v16bf ldfrag(const unsigned short* rowbase, int kh) {
    union { uint4 u[2]; v16bf v; } t;
    t.u[0] = *(const uint4*)(rowbase + kh * 8);
    t.u[1] = *(const uint4*)(rowbase + 16 + kh * 8);
    return t.v;
}

__global__ __launch_bounds__(256)
void qkv_rope_kernel(const float* __restrict__ x,
                     const float* __restrict__ W0, const float* __restrict__ W1,
                     const float* __restrict__ W2,
                     const float* __restrict__ b0, const float* __restrict__ b1,
                     const float* __restrict__ b2,
                     float* __restrict__ out) {
    __shared__ __align__(32) unsigned short As[BM * LDA];   // 20480 B
    __shared__ __align__(32) unsigned short Bs[BN * LDA];   // 40960 B

    const int t    = threadIdx.x;
    const int lane = t & 31;
    const int w    = t >> 5;          // 8 waves
    const int wm   = w & 1;           // 2 waves along M  (64 rows each)
    const int wn   = w >> 1;          // 4 waves along N  (64 cols each)
    const int lm   = lane & 15;
    const int lh   = lane >> 4;

    const int m0 = blockIdx.x * BM;
    const int n0 = blockIdx.y * BN;
    const int z  = blockIdx.z;        // 0=K, 1=Q, 2=V

    const float* W    = (z == 0) ? W0 : (z == 1) ? W1 : W2;
    const float* bias = (z == 0) ? b0 : (z == 1) ? b1 : b2;

    // ---- accumulators: 4 (M) x 4 (N) tiles of 16x16 f32 ----
    v8f acc[4][4];
#pragma unroll
    for (int mf = 0; mf < 4; ++mf)
#pragma unroll
        for (int nf = 0; nf < 4; ++nf)
#pragma unroll
            for (int e = 0; e < 8; ++e) acc[mf][nf][e] = 0.0f;

    // ---- staging thread mapping: r0 = row (stride 16), cg = float4 column ----
    const int r0 = t >> 4;            // 0..15
    const int cg = t & 15;            // 0..15 -> 16 float4 = 64 floats per row
    const float* xbase = x + (size_t)(m0 + r0) * DIM + cg * 4;
    const float* wbase = W + (size_t)(n0 + r0) * DIM + cg * 4;

    float4 ra[8], rb[16];
    // prologue: stage 0 -> regs
#pragma unroll
    for (int i = 0; i < 8; ++i)
        ra[i] = *(const float4*)(xbase + (size_t)i * 16 * DIM);
#pragma unroll
    for (int i = 0; i < 16; ++i)
        rb[i] = *(const float4*)(wbase + (size_t)i * 16 * DIM);

    for (int s = 0; s < NSTG; ++s) {
        // regs -> LDS (bf16)
#pragma unroll
        for (int i = 0; i < 8; ++i) {
            uint2 pa;
            pa.x = pack2(ra[i].x, ra[i].y);  pa.y = pack2(ra[i].z, ra[i].w);
            *(uint2*)(As + (r0 + 16 * i) * LDA + cg * 4) = pa;
        }
#pragma unroll
        for (int i = 0; i < 16; ++i) {
            uint2 pb;
            pb.x = pack2(rb[i].x, rb[i].y);  pb.y = pack2(rb[i].z, rb[i].w);
            *(uint2*)(Bs + (r0 + 16 * i) * LDA + cg * 4) = pb;
        }
        __syncthreads();

        // global -> regs for next stage (overlaps WMMA below)
        if (s + 1 < NSTG) {
            const float* xn = xbase + (size_t)(s + 1) * KC;
            const float* wn_ = wbase + (size_t)(s + 1) * KC;
#pragma unroll
            for (int i = 0; i < 8; ++i)
                ra[i] = *(const float4*)(xn + (size_t)i * 16 * DIM);
#pragma unroll
            for (int i = 0; i < 16; ++i)
                rb[i] = *(const float4*)(wn_ + (size_t)i * 16 * DIM);
            if (s + 2 < NSTG) {       // CDNA5 global_prefetch_b8 two stages ahead
                __builtin_prefetch(xbase + (size_t)(s + 2) * KC, 0, 1);
                __builtin_prefetch(wbase + (size_t)(s + 2) * KC, 0, 1);
            }
        }

        // ---- 2 k-steps x (4x4) WMMAs on this stage ----
#pragma unroll
        for (int k32 = 0; k32 < KC; k32 += 32) {
            v16bf af[4], bfr[4];
#pragma unroll
            for (int mf = 0; mf < 4; ++mf)
                af[mf] = ldfrag(As + (wm * 64 + mf * 16 + lm) * LDA + k32, lh);
#pragma unroll
            for (int nf = 0; nf < 4; ++nf)
                bfr[nf] = ldfrag(Bs + (wn * 64 + nf * 16 + lm) * LDA + k32, lh);
#pragma unroll
            for (int mf = 0; mf < 4; ++mf)
#pragma unroll
                for (int nf = 0; nf < 4; ++nf)
                    acc[mf][nf] = __builtin_amdgcn_wmma_f32_16x16x32_bf16(
                        false, af[mf], false, bfr[nf],
                        (short)0, acc[mf][nf], false, false);
        }
        __syncthreads();   // LDS reads done before next stage overwrites
    }

    // ---- epilogue: bias (+ RoPE for z<2), shuffle-based even/odd pairing ----
    float* outz = out + (size_t)z * (size_t)MTOT * DIM;
    const bool evenlane = (lm & 1) == 0;

    int    cgl[4];  float bcol[4];  size_t coff[4];
#pragma unroll
    for (int nf = 0; nf < 4; ++nf) {
        cgl[nf]  = n0 + wn * 64 + nf * 16 + lm;      // global projection column
        bcol[nf] = bias[cgl[nf]];
        int oc   = (evenlane ? cgl[nf] : (cgl[nf] - 1)) >> 1;
        coff[nf] = evenlane ? (size_t)oc : (size_t)(1024 + oc);
    }

    if (z < 2) {
#pragma unroll
        for (int mf = 0; mf < 4; ++mf) {
#pragma unroll
            for (int r = 0; r < 8; ++r) {
                const int m    = m0 + wm * 64 + mf * 16 + lh * 8 + r;
                const int sidx = m & (SEQ - 1);
                // theta = s * 10000^(-s/1024) = s * exp(-s * ln(10000)/1024)
                const float th = (float)sidx *
                                 __expf(-(float)sidx * (9.210340372f / 1024.0f));
                const float sn = __sinf(th), cs = __cosf(th);
#pragma unroll
                for (int nf = 0; nf < 4; ++nf) {
                    float v = acc[mf][nf][r] + bcol[nf];
                    float p = __shfl_xor(v, 1);          // even<->odd partner
                    float res = evenlane ? (v * cs - p * sn)    // x1*cos - x2*sin
                                         : (p * sn + v * cs);   // x1*sin + x2*cos
                    outz[(size_t)m * DIM + coff[nf]] = res;
                }
            }
        }
    } else {
#pragma unroll
        for (int mf = 0; mf < 4; ++mf)
#pragma unroll
            for (int r = 0; r < 8; ++r) {
                const int m = m0 + wm * 64 + mf * 16 + lh * 8 + r;
#pragma unroll
                for (int nf = 0; nf < 4; ++nf)
                    outz[(size_t)m * DIM + cgl[nf]] = acc[mf][nf][r] + bcol[nf];
            }
    }
}

extern "C" void kernel_launch(void* const* d_in, const int* in_sizes, int n_in,
                              void* d_out, int out_size, void* d_ws, size_t ws_size,
                              hipStream_t stream) {
    (void)in_sizes; (void)n_in; (void)out_size; (void)d_ws; (void)ws_size;
    const float* x  = (const float*)d_in[0];
    const float* Wk = (const float*)d_in[1];
    const float* bk = (const float*)d_in[2];
    const float* Wq = (const float*)d_in[3];
    const float* bq = (const float*)d_in[4];
    const float* Wv = (const float*)d_in[5];
    const float* bv = (const float*)d_in[6];
    float* out = (float*)d_out;

    dim3 grid(MTOT / BM, DIM / BN, 3);   // (128, 8, 3)
    dim3 block(256);                     // 8 waves (wave32)
    qkv_rope_kernel<<<grid, block, 0, stream>>>(x, Wk, Wq, Wv, bk, bq, bv, out);
}
